// SymphoniesDecoder_19018115186954
// MI455X (gfx1250) — compile-verified, matrix operand order checked by reading, not run on hardware
//
#include <hip/hip_runtime.h>
#include <math.h>

// ---------------- problem constants ----------------
#define C_    128
#define H_    8
#define DH_   16
#define NQK   100
#define XS_   128
#define YS_   128
#define ZS_   16
#define NVOX  (XS_*YS_*ZS_)     // 262144
#define NFOVK 32768
#define LF_   10080             // 48*160 + 24*80 + 12*40

typedef __attribute__((ext_vector_type(16))) _Float16 v16h;
typedef __attribute__((ext_vector_type(8)))  _Float16 v8h;
typedef __attribute__((ext_vector_type(8)))  float    v8f;

static __device__ __forceinline__ int clampi(int v, int lo, int hi) {
  return v < lo ? lo : (v > hi ? hi : v);
}

// pack two f32 -> packed 2xf16 dword (compiler emits cvt + pack)
static __device__ __forceinline__ unsigned pkh2(float a, float b) {
  union { _Float16 h[2]; unsigned u; } cv;
  cv.h[0] = (_Float16)a; cv.h[1] = (_Float16)b;
  return cv.u;
}

// K-permutation for fragment-order LDS storage of a 4-aligned K group:
// storage order [k0..7 | k16..23 | k8..15 | k24..31] so that one lane's whole
// WMMA fragment (ISA 7.12.2: half=0 -> K{0..7,16..23}, half=1 -> K{8..15,24..31})
// is 16 contiguous halfwords -> 2x ds_load_b128.
static __device__ __forceinline__ int kperm4(int c) {
  return c + (((c >> 3) & 1) << 3) - (((c >> 4) & 1) << 3);
}

// ============================================================
// GEMM: O[M,Nn] = act(A[M,K] @ W[K,Nn] + bias), f16 WMMA, f32 accumulate.
// Block tile 128x64, 256 threads = 8 waves; wave w owns rows [w*16, w*16+16)
// and all 4 N-subtiles: one A fragment feeds 4 v_wmma issues per K-step.
// Requires K%32==0, Nn%32==0 (true for every layer here). Out-of-range M/N
// handled by index clamping on loads + guards on stores only (garbage results
// are confined to unstored rows/columns), so staging is branch-free b128 loads.
// Double-buffered in registers: next tile's global loads issue before WMMAs.
// ============================================================
#define GBM 128
#define GBN 64
#define GBK 32

__launch_bounds__(256)
__global__ void k_gemm(const float* __restrict__ A, const float* __restrict__ W,
                       const float* __restrict__ bias, float* __restrict__ O,
                       int M, int K, int Nn, int relu)
{
  // 40-halfword row stride: 80 B, keeps half*16 offsets 16B-aligned and
  // spreads the 16 row-lanes over distinct 4-bank groups (stride 20 dwords).
  __shared__ _Float16 lsA[GBM][40];
  __shared__ _Float16 lsB[GBN][40];
  unsigned* lsA32 = (unsigned*)&lsA[0][0];   // row stride 20 dwords
  unsigned* lsB32 = (unsigned*)&lsB[0][0];

  const int tid  = threadIdx.x;
  const int wid  = tid >> 5, lane = tid & 31;
  const int mr   = lane & 15, half = lane >> 4;
  const int m0   = blockIdx.y * GBM;
  const int n0   = blockIdx.x * GBN;

  // A staging: 4 x float4 per thread; e = it*256+tid, row = e>>3, group = e&7
  // B staging: column n = tid&63, k-groups {kb, kb+16}, kb = (tid>>6)*4
  const int bn   = tid & 63;
  const int bkb  = (tid >> 6) * 4;
  const int gnc  = clampi(n0 + bn, 0, Nn - 1);

  float4 ra[4];
  float  rb[8];

  auto loadA = [&](int kk) {
    #pragma unroll
    for (int it = 0; it < 4; ++it) {
      int e = it * 256 + tid;
      int r = e >> 3, g = (e & 7) * 4;
      int gmc = clampi(m0 + r, 0, M - 1);
      ra[it] = *(const float4*)(A + (size_t)gmc * K + kk + g);
    }
  };
  auto loadB = [&](int kk) {
    #pragma unroll
    for (int h2 = 0; h2 < 2; ++h2) {
      int kr = kk + bkb + 16 * h2;
      #pragma unroll
      for (int j = 0; j < 4; ++j)
        rb[h2 * 4 + j] = W[(size_t)(kr + j) * Nn + gnc];
    }
  };
  auto storeA = [&]() {
    #pragma unroll
    for (int it = 0; it < 4; ++it) {
      int e = it * 256 + tid;
      int r = e >> 3, g = (e & 7) * 4;
      int s4 = kperm4(g);
      uint2 pk; pk.x = pkh2(ra[it].x, ra[it].y); pk.y = pkh2(ra[it].z, ra[it].w);
      *(uint2*)(lsA32 + r * 20 + (s4 >> 1)) = pk;
    }
  };
  auto storeB = [&]() {
    #pragma unroll
    for (int h2 = 0; h2 < 2; ++h2) {
      int s4 = kperm4(bkb + 16 * h2);
      uint2 pk; pk.x = pkh2(rb[h2*4+0], rb[h2*4+1]); pk.y = pkh2(rb[h2*4+2], rb[h2*4+3]);
      *(uint2*)(lsB32 + bn * 20 + (s4 >> 1)) = pk;
    }
  };

  v8f acc[4];
  #pragma unroll
  for (int t = 0; t < 4; ++t) acc[t] = (v8f){0.f,0.f,0.f,0.f,0.f,0.f,0.f,0.f};

  loadA(0); loadB(0);
  storeA(); storeB();
  __syncthreads();

  for (int k0 = 0; k0 < K; k0 += GBK) {
    const bool more = (k0 + GBK) < K;
    if (more) { loadA(k0 + GBK); loadB(k0 + GBK); }   // prefetch next tile

    union { v16h v; v8h h[2]; } af;
    const _Float16* pa = &lsA[wid * 16 + mr][half * 16];
    af.h[0] = *(const v8h*)pa;
    af.h[1] = *(const v8h*)(pa + 8);
    #pragma unroll
    for (int t = 0; t < 4; ++t) {
      union { v16h v; v8h h[2]; } bf;
      const _Float16* pb = &lsB[t * 16 + mr][half * 16];
      bf.h[0] = *(const v8h*)pb;
      bf.h[1] = *(const v8h*)(pb + 8);
      acc[t] = __builtin_amdgcn_wmma_f32_16x16x32_f16(
          false, af.v, false, bf.v, (short)0, acc[t], false, false);
    }
    __syncthreads();
    if (more) {
      storeA(); storeB();
      __syncthreads();
    }
  }

  // C/D layout: lane -> column (lane%16); VGPR i -> row i + 8*(lane/16)
  #pragma unroll
  for (int t = 0; t < 4; ++t) {
    int gn = n0 + t * 16 + mr;
    if (gn >= Nn) continue;
    float bv = bias ? bias[gn] : 0.f;
    #pragma unroll
    for (int i = 0; i < 8; ++i) {
      int gm = m0 + wid * 16 + i + 8 * half;
      if (gm < M) {
        float v = acc[t][i] + bv;
        if (relu) v = fmaxf(v, 0.f);
        O[(size_t)gm * Nn + gn] = v;
      }
    }
  }
}

// ---------------- LDS-tiled transpose: in[R,S] -> out[S,R] ----------------
__global__ void k_transpose(const float* __restrict__ in, float* __restrict__ out,
                            int R, int S)
{
  __shared__ float tile[32][33];
  int s0 = blockIdx.x * 32, r0 = blockIdx.y * 32;
  for (int j = threadIdx.y; j < 32; j += 8) {
    int r = r0 + j, s = s0 + threadIdx.x;
    tile[j][threadIdx.x] = (r < R && s < S) ? in[(size_t)r * S + s] : 0.f;
  }
  __syncthreads();
  for (int j = threadIdx.y; j < 32; j += 8) {
    int s = s0 + j, r = r0 + threadIdx.x;
    if (s < S && r < R) out[(size_t)s * R + r] = tile[threadIdx.x][j];
  }
}

// ---------------- elementwise helpers ----------------
__global__ void k_add(const float* __restrict__ a, const float* __restrict__ b,
                      float* __restrict__ o, int n)
{
  int i = blockIdx.x * 256 + threadIdx.x;
  if (i < n) o[i] = a[i] + b[i];
}

__global__ void k_copy(const float* __restrict__ a, float* __restrict__ o, int n)
{
  int i = blockIdx.x * 256 + threadIdx.x;
  if (i < n) o[i] = a[i];
}

// gather / scatter full C=128 rows by index
__global__ void k_gather_rows(const float* __restrict__ src, const int* __restrict__ idx,
                              float* __restrict__ dst)
{
  int i = blockIdx.x, c = threadIdx.x;
  dst[(size_t)i * C_ + c] = src[(size_t)idx[i] * C_ + c];
}
__global__ void k_scatter_rows(const float* __restrict__ src, const int* __restrict__ idx,
                               float* __restrict__ dst)
{
  int i = blockIdx.x, c = threadIdx.x;
  dst[(size_t)idx[i] * C_ + c] = src[(size_t)i * C_ + c];
}

// ---------------- fused residual + LayerNorm over C=128 ----------------
__launch_bounds__(128)
__global__ void k_ln_res(const float* __restrict__ a, const float* __restrict__ b,
                         const float* __restrict__ g, const float* __restrict__ beta,
                         float* __restrict__ out)
{
  __shared__ float red[128];
  int t = blockIdx.x, c = threadIdx.x;
  float x = a[(size_t)t * C_ + c] + (b ? b[(size_t)t * C_ + c] : 0.f);
  red[c] = x; __syncthreads();
  for (int s = 64; s > 0; s >>= 1) { if (c < s) red[c] += red[c + s]; __syncthreads(); }
  float mean = red[0] * (1.f / 128.f); __syncthreads();
  float d = x - mean;
  red[c] = d * d; __syncthreads();
  for (int s = 64; s > 0; s >>= 1) { if (c < s) red[c] += red[c + s]; __syncthreads(); }
  float var = red[0] * (1.f / 128.f);
  out[(size_t)t * C_ + c] = d * rsqrtf(var + 1e-5f) * g[c] + beta[c];
}

// ---------------- MHA core: online softmax, K/V staged in LDS ----------------
// heads H=8, Dh=16, head-interleaved layout [tok, h*16+d]. scale 1/sqrt(16).
__launch_bounds__(128)
__global__ void k_mha(const float* __restrict__ Qp, const float* __restrict__ Kp,
                      const float* __restrict__ Vp, float* __restrict__ O,
                      int M, int Nk)
{
  __shared__ float sK[NQK * 16];
  __shared__ float sV[NQK * 16];
  const int h = blockIdx.y, tid = threadIdx.x;
  for (int e = tid; e < Nk * 16; e += 128) {
    int kk = e >> 4, d = e & 15;
    sK[e] = Kp[(size_t)kk * C_ + h * 16 + d];
    sV[e] = Vp[(size_t)kk * C_ + h * 16 + d];
  }
  __syncthreads();
  int m = blockIdx.x * 128 + tid;
  if (m >= M) return;
  float q[16], acc[16];
  #pragma unroll
  for (int d = 0; d < 16; ++d) { q[d] = Qp[(size_t)m * C_ + h * 16 + d]; acc[d] = 0.f; }
  float mx = -1e30f, sum = 0.f;
  for (int kk = 0; kk < Nk; ++kk) {
    float s = 0.f;
    #pragma unroll
    for (int d = 0; d < 16; ++d) s += q[d] * sK[kk * 16 + d];
    s *= 0.25f;
    float nm = fmaxf(mx, s);
    float corr = __expf(mx - nm);
    float w = __expf(s - nm);
    sum = sum * corr + w;
    #pragma unroll
    for (int d = 0; d < 16; ++d) acc[d] = acc[d] * corr + w * sV[kk * 16 + d];
    mx = nm;
  }
  float inv = 1.f / sum;
  #pragma unroll
  for (int d = 0; d < 16; ++d) O[(size_t)m * C_ + h * 16 + d] = acc[d] * inv;
}

// ---------------- deformable 3D sampling (P=8, L=1) ----------------
// thread = (head h = tid>>4, channel d = tid&15); the 16 d-lanes of a head read
// one contiguous 64B line per trilinear corner -> gathers served from L2
// (the 134MB projected-value tensor fits in the 192MB L2).
__launch_bounds__(128)
__global__ void k_deform3d(const float* __restrict__ V, const float* __restrict__ OFF,
                           const float* __restrict__ WGT, const float* __restrict__ REF,
                           const int* __restrict__ qidx, float* __restrict__ O, int M)
{
  const int q = blockIdx.x; if (q >= M) return;
  const int tid = threadIdx.x;
  const int h = tid >> 4, d = tid & 15;
  const int rr = qidx ? qidx[q] : q;
  const float rx = REF[(size_t)rr*3+0], ry = REF[(size_t)rr*3+1], rz = REF[(size_t)rr*3+2];
  float wr[8], wmax = -1e30f;
  #pragma unroll
  for (int p = 0; p < 8; ++p) { wr[p] = WGT[(size_t)q*64 + h*8 + p]; wmax = fmaxf(wmax, wr[p]); }
  float wsum = 0.f;
  #pragma unroll
  for (int p = 0; p < 8; ++p) { wr[p] = __expf(wr[p] - wmax); wsum += wr[p]; }
  const float winv = 1.f / wsum;
  float acc = 0.f;
  for (int p = 0; p < 8; ++p) {
    const float* ob = OFF + (size_t)q*192 + (size_t)(h*8 + p)*3;
    float cx = (rx + ob[0] * (1.f/128.f)) * 128.f - 0.5f;
    float cy = (ry + ob[1] * (1.f/128.f)) * 128.f - 0.5f;
    float cz = (rz + ob[2] * (1.f/16.f))  * 16.f  - 0.5f;
    float x0 = floorf(cx), y0 = floorf(cy), z0 = floorf(cz);
    float ws = wr[p] * winv;
    #pragma unroll
    for (int dx = 0; dx < 2; ++dx)
    #pragma unroll
    for (int dy = 0; dy < 2; ++dy)
    #pragma unroll
    for (int dz = 0; dz < 2; ++dz) {
      float xi = x0 + dx, yi = y0 + dy, zi = z0 + dz;
      float wt = (1.f - fabsf(cx - xi)) * (1.f - fabsf(cy - yi)) * (1.f - fabsf(cz - zi));
      bool ok = (xi >= 0.f) & (xi < 128.f) & (yi >= 0.f) & (yi < 128.f) & (zi >= 0.f) & (zi < 16.f);
      int ix = clampi((int)xi, 0, XS_-1);
      int iy = clampi((int)yi, 0, YS_-1);
      int iz = clampi((int)zi, 0, ZS_-1);
      int n = (ix * YS_ + iy) * ZS_ + iz;
      float g = V[(size_t)n * C_ + h * 16 + d];
      acc += g * (ok ? wt : 0.f) * ws;
    }
  }
  O[(size_t)q * C_ + h * 16 + d] = acc;
}

// ---------------- deformable 2D sampling (L=3 pyramid levels, P=4) ----------------
__launch_bounds__(128)
__global__ void k_deform2d(const float* __restrict__ V, const float* __restrict__ OFF,
                           const float* __restrict__ WGT, const float* __restrict__ REF,
                           float* __restrict__ O, int M)
{
  const int q = blockIdx.x; if (q >= M) return;
  const int tid = threadIdx.x;
  const int h = tid >> 4, d = tid & 15;
  const int lvlH[3] = {48, 24, 12};
  const int lvlW[3] = {160, 80, 40};
  const int lvlS[3] = {0, 7680, 9600};
  float wr[12], wmax = -1e30f;
  #pragma unroll
  for (int i = 0; i < 12; ++i) { wr[i] = WGT[(size_t)q*96 + h*12 + i]; wmax = fmaxf(wmax, wr[i]); }
  float wsum = 0.f;
  #pragma unroll
  for (int i = 0; i < 12; ++i) { wr[i] = __expf(wr[i] - wmax); wsum += wr[i]; }
  const float winv = 1.f / wsum;
  float acc = 0.f;
  for (int l = 0; l < 3; ++l) {
    const float hh = (float)lvlH[l], ww = (float)lvlW[l];
    const float rx = REF[(size_t)q*6 + l*2 + 0];
    const float ry = REF[(size_t)q*6 + l*2 + 1];
    for (int p = 0; p < 4; ++p) {
      const float* ob = OFF + (size_t)q*192 + (size_t)((h*3 + l)*4 + p)*2;
      float cx = (rx + ob[0] / ww) * ww - 0.5f;
      float cy = (ry + ob[1] / hh) * hh - 0.5f;
      float x0 = floorf(cx), y0 = floorf(cy);
      float ws = wr[l*4 + p] * winv;
      #pragma unroll
      for (int dx = 0; dx < 2; ++dx)
      #pragma unroll
      for (int dy = 0; dy < 2; ++dy) {
        float xi = x0 + dx, yi = y0 + dy;
        float wt = (1.f - fabsf(cx - xi)) * (1.f - fabsf(cy - yi));
        bool ok = (xi >= 0.f) & (xi < ww) & (yi >= 0.f) & (yi < hh);
        int ix = clampi((int)xi, 0, lvlW[l]-1);
        int iy = clampi((int)yi, 0, lvlH[l]-1);
        int n = lvlS[l] + iy * lvlW[l] + ix;
        float g = V[(size_t)n * C_ + h * 16 + d];
        acc += g * (ok ? wt : 0.f) * ws;
      }
    }
  }
  O[(size_t)q * C_ + h * 16 + d] = acc;
}

// ============================================================
// host side
// ============================================================

// param leaf sizes in jax pytree (sorted-dict) flatten order
static const int PSZ[68] = {
  // l1: attn(off,out,val,wgt each b,w), ffn(w1,w2), ln1(b,g), ln2(b,g)
  192, 128*192, 128, 128*128, 128, 128*128, 96, 128*96,
  512, 128*512, 128, 512*128, 128, 128, 128, 128,
  // l2: ln1(b,g), mha(k,o,q,v each b,w)
  128, 128, 128, 128*128, 128, 128*128, 128, 128*128, 128, 128*128,
  // l3: attn, ffn, ln1, ln2
  192, 128*192, 128, 128*128, 128, 128*128, 64, 128*64,
  512, 128*512, 128, 512*128, 128, 128, 128, 128,
  // l4: attn, ln1
  192, 128*192, 128, 128*128, 128, 128*128, 64, 128*64, 128, 128,
  // l5: ffn, ln1, ln2, mha(k,o,q,v)
  512, 128*512, 128, 512*128, 128, 128, 128, 128,
  128, 128*128, 128, 128*128, 128, 128*128, 128, 128*128
};
// leaf indices
enum {
  L1_OFF_B=0, L1_OFF_W, L1_OUT_B, L1_OUT_W, L1_VAL_B, L1_VAL_W, L1_WGT_B, L1_WGT_W,
  L1_W1_B, L1_W1_W, L1_W2_B, L1_W2_W, L1_LN1_B, L1_LN1_G, L1_LN2_B, L1_LN2_G,
  L2_LN1_B, L2_LN1_G, L2_K_B, L2_K_W, L2_O_B, L2_O_W, L2_Q_B, L2_Q_W, L2_V_B, L2_V_W,
  L3_OFF_B, L3_OFF_W, L3_OUT_B, L3_OUT_W, L3_VAL_B, L3_VAL_W, L3_WGT_B, L3_WGT_W,
  L3_W1_B, L3_W1_W, L3_W2_B, L3_W2_W, L3_LN1_B, L3_LN1_G, L3_LN2_B, L3_LN2_G,
  L4_OFF_B, L4_OFF_W, L4_OUT_B, L4_OUT_W, L4_VAL_B, L4_VAL_W, L4_WGT_B, L4_WGT_W,
  L4_LN1_B, L4_LN1_G,
  L5_W1_B, L5_W1_W, L5_W2_B, L5_W2_W, L5_LN1_B, L5_LN1_G, L5_LN2_B, L5_LN2_G,
  L5_K_B, L5_K_W, L5_O_B, L5_O_W, L5_Q_B, L5_Q_W, L5_V_B, L5_V_W
};

extern "C" void kernel_launch(void* const* d_in, const int* in_sizes, int n_in,
                              void* d_out, int out_size, void* d_ws, size_t ws_size,
                              hipStream_t stream)
{
  (void)in_sizes; (void)out_size; (void)ws_size;

  const float* scene_embed = (const float*)d_in[0];   // (C, N) c-major
  const float* scene_pos   = (const float*)d_in[1];   // (C, N)
  const float* inst_q      = (const float*)d_in[2];   // (NQ, C)
  const float* inst_pos    = (const float*)d_in[3];   // (NQ, C)
  const float* feat0       = (const float*)d_in[4];   // (C, 48*160)
  const float* feat1       = (const float*)d_in[5];   // (C, 24*80)
  const float* feat2       = (const float*)d_in[6];   // (C, 12*40)
  const float* ref2d       = (const float*)d_in[7];   // (NQ, 3, 2)
  const float* ref3d       = (const float*)d_in[8];   // (NQ, 1, 3)
  const float* refvox      = (const float*)d_in[9];   // (N, 1, 3)
  const int*   fov_idx     = (const int*)  d_in[10];  // (NFOV,)

  // params: either 68 separate leaves (d_in[11..78]) or one flat buffer d_in[11]
  const float* P[68];
  if (n_in >= 11 + 68) {
    for (int i = 0; i < 68; ++i) P[i] = (const float*)d_in[11 + i];
  } else {
    const float* base = (const float*)d_in[11];
    size_t off = 0;
    for (int i = 0; i < 68; ++i) { P[i] = base + off; off += (size_t)PSZ[i]; }
  }

  float* out = (float*)d_out;

  // ---- workspace layout ----
  float* w = (float*)d_ws;
  size_t off = 0;
  auto alloc = [&](size_t n) { float* p = w + off; off += n; return p; };
  float* SF    = alloc((size_t)NVOX * C_);   // scene token-major
  float* VV    = alloc((size_t)NVOX * C_);   // value projection (reused l3/l4)
  float* FOV0  = alloc((size_t)NFOVK * C_);  // fov state (progressively updated)
  float* FOVP  = alloc((size_t)NFOVK * C_);
  float* FQ    = alloc((size_t)NFOVK * C_);
  float* FT1   = alloc((size_t)NFOVK * C_);
  float* FT2   = alloc((size_t)NFOVK * C_);
  float* FH    = alloc((size_t)NFOVK * 512); // ffn hidden
  float* DOFF  = alloc((size_t)NFOVK * 192);
  float* DWGT  = alloc((size_t)NFOVK * 64);
  float* FEATS = alloc((size_t)LF_ * C_);
  float* VFEAT = alloc((size_t)LF_ * C_);
  float* IB    = alloc((size_t)NQK * C_);    // inst state
  float* IT1   = alloc((size_t)NQK * 512);
  float* IT2   = alloc((size_t)NQK * 512);
  float* IT3   = alloc((size_t)NQK * 512);
  float* IT4   = alloc((size_t)NQK * 512);
  float* IH    = alloc((size_t)NQK * 512);
  float* IOFF  = alloc((size_t)NQK * 192);
  float* IWGT  = alloc((size_t)NQK * 96);

  auto gemm = [&](const float* A, const float* W_, const float* b, float* O,
                  int M, int K, int Nn, int relu) {
    dim3 g((Nn + GBN - 1) / GBN, (M + GBM - 1) / GBM);
    k_gemm<<<g, 256, 0, stream>>>(A, W_, b, O, M, K, Nn, relu);
  };
  auto transpose = [&](const float* in, float* o, int R, int S) {
    dim3 g((S + 31) / 32, (R + 31) / 32);
    k_transpose<<<g, dim3(32, 8), 0, stream>>>(in, o, R, S);
  };
  auto add = [&](const float* a, const float* b, float* o, int n) {
    k_add<<<(n + 255) / 256, 256, 0, stream>>>(a, b, o, n);
  };
  auto ln = [&](const float* a, const float* b, int g_idx, int b_idx, float* o, int M) {
    k_ln_res<<<M, 128, 0, stream>>>(a, b, P[g_idx], P[b_idx], o);
  };

  // ================= stage A: layer l1 (instance deformable 2D) =================
  transpose(feat0, FEATS + (size_t)0    * C_, C_, 48 * 160);
  transpose(feat1, FEATS + (size_t)7680 * C_, C_, 24 * 80);
  transpose(feat2, FEATS + (size_t)9600 * C_, C_, 12 * 40);
  add(inst_q, inst_pos, IT1, NQK * C_);                                      // q1
  gemm(FEATS, P[L1_VAL_W], P[L1_VAL_B], VFEAT, LF_, C_, C_, 0);              // value proj
  gemm(IT1, P[L1_OFF_W], P[L1_OFF_B], IOFF, NQK, C_, 192, 0);
  gemm(IT1, P[L1_WGT_W], P[L1_WGT_B], IWGT, NQK, C_, 96, 0);
  k_deform2d<<<NQK, 128, 0, stream>>>(VFEAT, IOFF, IWGT, ref2d, IT2, NQK);
  gemm(IT2, P[L1_OUT_W], P[L1_OUT_B], IT3, NQK, C_, C_, 0);
  ln(inst_q, IT3, L1_LN1_G, L1_LN1_B, IB, NQK);                              // inst_a
  gemm(IB, P[L1_W1_W], P[L1_W1_B], IH, NQK, C_, 512, 1);
  gemm(IH, P[L1_W2_W], P[L1_W2_B], IT2, NQK, 512, C_, 0);
  ln(IB, IT2, L1_LN2_G, L1_LN2_B, IB, NQK);                                  // inst after l1

  // ================= scene flatten + fov gather =================
  transpose(scene_embed, SF, C_, NVOX);             // SF = (N, C)
  transpose(scene_pos,   VV, C_, NVOX);             // VV temp = pos token-major
  k_gather_rows<<<NFOVK, 128, 0, stream>>>(VV, fov_idx, FOVP);
  k_gather_rows<<<NFOVK, 128, 0, stream>>>(SF, fov_idx, FOV0);

  // ================= stage B: layer l2 (fov <- inst cross-MHA) =================
  add(FOV0, FOVP, FQ, NFOVK * C_);                                           // q2
  add(IB, inst_pos, IT1, NQK * C_);                                          // k input
  gemm(FQ,  P[L2_Q_W], P[L2_Q_B], FT1, NFOVK, C_, C_, 0);                    // Qp
  gemm(IT1, P[L2_K_W], P[L2_K_B], IT2, NQK, C_, C_, 0);                      // Kp
  gemm(IB,  P[L2_V_W], P[L2_V_B], IT3, NQK, C_, C_, 0);                      // Vp
  k_mha<<<dim3(NFOVK / 128, H_), 128, 0, stream>>>(FT1, IT2, IT3, FT2, NFOVK, NQK);
  gemm(FT2, P[L2_O_W], P[L2_O_B], FT1, NFOVK, C_, C_, 0);
  ln(FOV0, FT1, L2_LN1_G, L2_LN1_B, FOV0, NFOVK);                            // fov_a

  // ================= stage C: layer l3 (fov deformable 3D + FFN) =================
  gemm(SF, P[L3_VAL_W], P[L3_VAL_B], VV, NVOX, C_, C_, 0);                   // big val proj
  add(FOV0, FOVP, FQ, NFOVK * C_);                                           // q3
  gemm(FQ, P[L3_OFF_W], P[L3_OFF_B], DOFF, NFOVK, C_, 192, 0);
  gemm(FQ, P[L3_WGT_W], P[L3_WGT_B], DWGT, NFOVK, C_, 64, 0);
  k_deform3d<<<NFOVK, 128, 0, stream>>>(VV, DOFF, DWGT, refvox, fov_idx, FT1, NFOVK);
  gemm(FT1, P[L3_OUT_W], P[L3_OUT_B], FT2, NFOVK, C_, C_, 0);
  ln(FOV0, FT2, L3_LN1_G, L3_LN1_B, FOV0, NFOVK);                            // fov_b
  gemm(FOV0, P[L3_W1_W], P[L3_W1_B], FH, NFOVK, C_, 512, 1);
  gemm(FH, P[L3_W2_W], P[L3_W2_B], FT1, NFOVK, 512, C_, 0);
  ln(FOV0, FT1, L3_LN2_G, L3_LN2_B, FOV0, NFOVK);                            // fov_c
  k_scatter_rows<<<NFOVK, 128, 0, stream>>>(FOV0, fov_idx, SF);              // update scene

  // ================= stage D: layer l4 (inst deformable 3D over updated scene) =====
  gemm(SF, P[L4_VAL_W], P[L4_VAL_B], VV, NVOX, C_, C_, 0);                   // big val proj 2
  add(IB, inst_pos, IT1, NQK * C_);
  gemm(IT1, P[L4_OFF_W], P[L4_OFF_B], IOFF, NQK, C_, 192, 0);
  gemm(IT1, P[L4_WGT_W], P[L4_WGT_B], IWGT, NQK, C_, 64, 0);
  k_deform3d<<<NQK, 128, 0, stream>>>(VV, IOFF, IWGT, ref3d, (const int*)nullptr, IT2, NQK);
  gemm(IT2, P[L4_OUT_W], P[L4_OUT_B], IT3, NQK, C_, C_, 0);
  ln(IB, IT3, L4_LN1_G, L4_LN1_B, IB, NQK);                                  // inst_c

  // ================= stage E: layer l5 (inst self-MHA + FFN) =================
  add(IB, inst_pos, IT1, NQK * C_);                                          // q = k input
  gemm(IT1, P[L5_Q_W], P[L5_Q_B], IT2, NQK, C_, C_, 0);
  gemm(IT1, P[L5_K_W], P[L5_K_B], IT3, NQK, C_, C_, 0);
  gemm(IB,  P[L5_V_W], P[L5_V_B], IT4, NQK, C_, C_, 0);
  k_mha<<<dim3((NQK + 127) / 128, H_), 128, 0, stream>>>(IT2, IT3, IT4, IT1, NQK, NQK);
  gemm(IT1, P[L5_O_W], P[L5_O_B], IT2, NQK, C_, C_, 0);
  ln(IB, IT2, L5_LN1_G, L5_LN1_B, IB, NQK);
  gemm(IB, P[L5_W1_W], P[L5_W1_B], IH, NQK, C_, 512, 1);
  gemm(IH, P[L5_W2_W], P[L5_W2_B], IT1, NQK, 512, C_, 0);
  ln(IB, IT1, L5_LN2_G, L5_LN2_B, IB, NQK);                                  // final inst

  // ================= outputs =================
  transpose(SF, out, NVOX, C_);                      // scene_out (C, N)
  k_copy<<<(NQK * C_ + 255) / 256, 256, 0, stream>>>(IB, out + (size_t)C_ * NVOX, NQK * C_);
}
